// HandwritingPredictionNetwork_18992345383357
// MI455X (gfx1250) — matrix operand order repaired and checked by reading, not compiled, not run on hardware
//
#include <hip/hip_runtime.h>
#include <hip/hip_bf16.h>

// ---------------------------------------------------------------------------
// HandwritingPredictionNetwork for MI455X (gfx1250, wave32, WMMA)
//
// B=64, T=1024, I=3, H=900, K=20.
// LSTM recurrence: per-step [64,903]@[903,3600] in f16 WMMA (f32 accum).
// Weight B-fragments are loop-invariant per wave -> compiler promotes all 29
// (232 VGPRs) into the extended VGPR file (s_set_vgpr_msb) and keeps them
// resident across the whole 1024-step recurrence. A-fragments stream through
// a 4-slot register ring with sched_barriers pinning load-ahead-of-WMMA so
// each v_wmma waits only on a 3-iteration-old load (s_wait_loadcnt 6, not 0).
// Head GEMM [65536,928]@[928,128] WMMA, fully parallel, same pipelining.
// ---------------------------------------------------------------------------

#define B_    64
#define T_    1024
#define IN_   3
#define H_    900
#define K4_   3600          // 4*H
#define KTOT_ (IN_ + H_)    // 903 real K
#define KP_   928           // K padded to 29 tiles of 32
#define KT_   29            // k tiles
#define NT_   225           // n tiles (3600/16) for LSTM gemm
#define HNP_  128           // head outputs padded (121 -> 128)
#define HNT_  8             // head n tiles
#define BT_   (B_ * T_)     // 65536 rows for head gemm
#define RT_   (BT_ / 16)    // 4096 row tiles
#define PIPE_ 3             // load-pairs in flight ahead of the consuming wmma

typedef __attribute__((ext_vector_type(16))) _Float16 v16h;
typedef __attribute__((ext_vector_type(8)))  float    v8f;

union FragU { unsigned u[8]; uint4 q[2]; v16h h; };
union PackU { _Float16 h[2]; unsigned u; };

#if __has_builtin(__builtin_amdgcn_sched_barrier)
#define SCHED_FENCE() __builtin_amdgcn_sched_barrier(0)
#else
#define SCHED_FENCE() asm volatile("" ::: "memory")
#endif

// ---------------------------------------------------------------------------
// A-matrix fragment layout (16x32 f16, ISA 7.12.2):
//   lane L: row m = L%16, K-half = L/16
//   VGPR j (0..7): k = (j/4)*16 + (L/16)*8 + (j%4)*2   (pair k, k+1)
// Fragments stored lane-major: dword index = frag*256 + lane*8 + j
// -> a wave reads one fragment with two global_load_b128 per lane.
// ---------------------------------------------------------------------------
__device__ __forceinline__ int afrag_idx(int row, int k) {
  int mt = row >> 4, mr = row & 15;
  int kt = k >> 5,  kin = k & 31;
  int g = kin >> 4, rem = kin & 15;
  int half = rem >> 3, off = rem & 7;
  int j    = (g << 2) + (off >> 1);
  int lane = (half << 4) + mr;
  return ((((mt * KT_ + kt) << 8) + lane * 8 + j) << 1) + (off & 1);
}

__device__ __forceinline__ float sigmoidf_(float v) {
  return 1.0f / (1.0f + __expf(-v));
}

__device__ __forceinline__ void load_frag(FragU& f, const unsigned* p) {
  f.q[0] = ((const uint4*)p)[0];
  f.q[1] = ((const uint4*)p)[1];
}

// ---------------------------------------------------------------------------
// Pre-swizzle W_xh [903, 3600] f32 -> f16 B-fragments (32x16 per tile).
// B layout: lane L holds col n (=L%16), VGPR j rows k=(j/4)*16+(L/16)*8+(j%4)*2
// (pair k, k+1), matching the A layout.  K rows >= 903 are zero padding.
// ---------------------------------------------------------------------------
__global__ void prep_w_kernel(const float* __restrict__ W, unsigned* __restrict__ wf) {
  int tid = blockIdx.x * blockDim.x + threadIdx.x;
  if (tid >= NT_ * KT_ * 256) return;
  int j = tid & 7, l = (tid >> 3) & 31, frag = tid >> 8;
  int kt = frag % KT_, nt = frag / KT_;
  int n  = nt * 16 + (l & 15);
  int kk = kt * 32 + ((j >> 2) << 4) + ((l >> 4) << 3) + ((j & 3) << 1);
  float lo = (kk     < KTOT_) ? W[(size_t)kk       * K4_ + n] : 0.0f;
  float hi = (kk + 1 < KTOT_) ? W[(size_t)(kk + 1) * K4_ + n] : 0.0f;
  PackU p; p.h[0] = (_Float16)lo; p.h[1] = (_Float16)hi;
  wf[tid] = p.u;
}

// Combined head weight matrix: col n in [0,121): pi(20)|mu(40)|sd(40)|ro(20)|eos(1)
__device__ __forceinline__ float head_w(int n, int kk,
    const float* pw, const float* mw, const float* sw, const float* rw, const float* ew) {
  if (kk >= H_) return 0.0f;
  if (n < 20)   return pw[n * H_ + kk];
  if (n < 60)   return mw[(n - 20) * H_ + kk];
  if (n < 100)  return sw[(n - 60) * H_ + kk];
  if (n < 120)  return rw[(n - 100) * H_ + kk];
  if (n == 120) return ew[kk];
  return 0.0f;
}

__global__ void prep_head_kernel(const float* __restrict__ pw, const float* __restrict__ mw,
                                 const float* __restrict__ sw, const float* __restrict__ rw,
                                 const float* __restrict__ ew, unsigned* __restrict__ hwf) {
  int tid = blockIdx.x * blockDim.x + threadIdx.x;
  if (tid >= HNT_ * KT_ * 256) return;
  int j = tid & 7, l = (tid >> 3) & 31, frag = tid >> 8;
  int kt = frag % KT_, nt = frag / KT_;
  int n  = nt * 16 + (l & 15);
  int kk = kt * 32 + ((j >> 2) << 4) + ((l >> 4) << 3) + ((j & 3) << 1);
  PackU p;
  p.h[0] = (_Float16)head_w(n, kk,     pw, mw, sw, rw, ew);
  p.h[1] = (_Float16)head_w(n, kk + 1, pw, mw, sw, rw, ew);
  hwf[tid] = p.u;
}

// Zero initial state: A fragments (h0=0 + K padding), c0, hidden-seq
// fragments (so K padding 900..927 is zero), and the barrier counter.
__global__ void init_kernel(unsigned* __restrict__ af, float* __restrict__ c,
                            unsigned* __restrict__ hsf, unsigned* __restrict__ cnt) {
  long tid = (long)blockIdx.x * blockDim.x + threadIdx.x;
  long nth = (long)gridDim.x * blockDim.x;
  for (long i = tid; i < (long)4 * KT_ * 256; i += nth)   af[i]  = 0u;
  for (long i = tid; i < (long)B_ * H_; i += nth)         c[i]   = 0.0f;
  for (long i = tid; i < (long)RT_ * KT_ * 256; i += nth) hsf[i] = 0u;
  if (tid == 0) *cnt = 0u;
}

// Write x_{t=0} into A-fragment columns 0..2 (after init zeroing).
__global__ void fill_x0_kernel(const float* __restrict__ x, unsigned* __restrict__ af) {
  int tid = threadIdx.x;
  if (tid >= B_ * IN_) return;
  int b = tid / IN_, i = tid - b * IN_;
  ((_Float16*)af)[afrag_idx(b, i)] = (_Float16)x[(size_t)(b * T_) * IN_ + i];
}

// ---------------------------------------------------------------------------
// Global barrier (monotonic counter + acquire spin). 225 blocks x 4 waves
// is trivially co-resident on MI455X (wave32, >200 WGPs).
// ---------------------------------------------------------------------------
__device__ __forceinline__ void grid_sync(unsigned* cnt, unsigned target) {
  __threadfence();
  __syncthreads();
  if (threadIdx.x == 0) {
    __hip_atomic_fetch_add(cnt, 1u, __ATOMIC_RELEASE, __HIP_MEMORY_SCOPE_AGENT);
    while (__hip_atomic_load(cnt, __ATOMIC_ACQUIRE, __HIP_MEMORY_SCOPE_AGENT) < target)
      __builtin_amdgcn_s_sleep(2);
  }
  __syncthreads();
}

// ---------------------------------------------------------------------------
// Persistent LSTM kernel. Block = n-tile (225 blocks), wave = m-tile (4 waves).
// Phase 1: z[64,3600] tile via 29 x v_wmma_f32_16x16x32_f16, A-side streamed
//          through a 4-slot ring with PIPE_=3 load-pairs in flight.
// Phase 2: peephole gates + state update; h_new stored as f16 directly in
//          A-fragment layout (next step) and hidden-seq fragment layout.
// ---------------------------------------------------------------------------
__global__ __launch_bounds__(128) void lstm_persistent(
    const float* __restrict__ x, const unsigned* __restrict__ wf,
    const float* __restrict__ W_ci, const float* __restrict__ b_i,
    const float* __restrict__ W_cf, const float* __restrict__ b_f,
    const float* __restrict__ b_c,
    const float* __restrict__ W_co, const float* __restrict__ b_o,
    unsigned* __restrict__ af, float* __restrict__ z, float* __restrict__ c,
    unsigned* __restrict__ hsf, unsigned* __restrict__ cnt) {

  const int nt   = blockIdx.x;          // 0..224
  const int mt   = threadIdx.x >> 5;    // wave id == m tile (0..3)
  const int lane = threadIdx.x & 31;
  const int tid  = blockIdx.x * blockDim.x + threadIdx.x;
  const int nth  = gridDim.x * blockDim.x;
  _Float16* afh = (_Float16*)af;
  _Float16* hsh = (_Float16*)hsf;

  const unsigned* abase = af + (((size_t)mt * KT_) << 8) + lane * 8;
  const unsigned* bbase = wf + (((size_t)nt * KT_) << 8) + lane * 8;

  unsigned phase = 0;
  for (int t = 0; t < T_; ++t) {
    // ---- phase 1: GEMM tile: z[mt*16 .. , nt*16 ..] ----
    {
      v8f acc = {};
      FragU a[PIPE_ + 1];
      #pragma unroll
      for (int p = 0; p < PIPE_; ++p)               // prime the pipeline
        load_frag(a[p], abase + (p << 8));
      SCHED_FENCE();
      #pragma unroll
      for (int kt = 0; kt < KT_; ++kt) {
        const int cur = kt & PIPE_;                 // PIPE_+1 == 4 slots, mask 3
        if (kt + PIPE_ < KT_)
          load_frag(a[(kt + PIPE_) & PIPE_], abase + ((kt + PIPE_) << 8));
        FragU b;                                    // loop-invariant across t ->
        load_frag(b, bbase + (kt << 8));            // hoisted into resident VGPRs
        SCHED_FENCE();                              // keep load-ahead ordering
        acc = __builtin_amdgcn_wmma_f32_16x16x32_f16(
            false, a[cur].h, false, b.h, (short)0, acc, false, false);
      }
      const int col   = nt * 16 + (lane & 15);
      const int rbase = mt * 16 + ((lane >> 4) << 3);
      #pragma unroll
      for (int r = 0; r < 8; ++r) z[(size_t)(rbase + r) * K4_ + col] = acc[r];
    }
    ++phase; grid_sync(cnt, phase * gridDim.x);

    // ---- phase 2: peephole gates ----
    for (int i = tid; i < B_ * H_; i += nth) {
      int b = i / H_, hh = i - b * H_;
      const float* zr = z + (size_t)b * K4_;
      float zi = zr[hh], zf = zr[H_ + hh], zc = zr[2 * H_ + hh], zo = zr[3 * H_ + hh];
      float cp = c[i];
      float it = sigmoidf_(zi + cp * W_ci[hh] + b_i[hh]);
      float ft = sigmoidf_(zf + cp * W_cf[hh] + b_f[hh]);
      float cn = ft * cp + it * tanhf(zc + b_c[hh]);
      float ot = sigmoidf_(zo + cn * W_co[hh] + b_o[hh]);
      float hn = ot * tanhf(cn);
      c[i] = cn;
      _Float16 hf = (_Float16)hn;
      afh[afrag_idx(b, IN_ + hh)] = hf;          // next-step A operand
      hsh[afrag_idx(b * T_ + t, hh)] = hf;       // hidden seq (head A operand)
    }
    if (t + 1 < T_) {
      for (int i = tid; i < B_ * IN_; i += nth) {
        int b = i / IN_, ii = i - b * IN_;
        afh[afrag_idx(b, ii)] = (_Float16)x[((size_t)b * T_ + (t + 1)) * IN_ + ii];
      }
    }
    ++phase; grid_sync(cnt, phase * gridDim.x);
  }
}

// ---------------------------------------------------------------------------
// Head GEMM: raw[65536,128] = hs[65536,928] @ Whead[928,128]; 1 tile / wave,
// both operand streams pipelined through 4-slot rings.
// ---------------------------------------------------------------------------
__global__ __launch_bounds__(256) void head_matmul_kernel(
    const unsigned* __restrict__ hsf, const unsigned* __restrict__ hwf,
    float* __restrict__ raw) {
  int gw   = (blockIdx.x * blockDim.x + threadIdx.x) >> 5;
  int lane = threadIdx.x & 31;
  int rt = gw >> 3, nt = gw & 7;
  if (rt >= RT_) return;

  const unsigned* abase = hsf + (((size_t)rt * KT_) << 8) + lane * 8;
  const unsigned* bbase = hwf + (((size_t)nt * KT_) << 8) + lane * 8;

  v8f acc = {};
  FragU a[PIPE_ + 1], b[PIPE_ + 1];
  #pragma unroll
  for (int p = 0; p < PIPE_; ++p) {
    load_frag(a[p], abase + (p << 8));
    load_frag(b[p], bbase + (p << 8));
  }
  SCHED_FENCE();
  #pragma unroll
  for (int kt = 0; kt < KT_; ++kt) {
    const int cur = kt & PIPE_;
    if (kt + PIPE_ < KT_) {
      load_frag(a[(kt + PIPE_) & PIPE_], abase + ((kt + PIPE_) << 8));
      load_frag(b[(kt + PIPE_) & PIPE_], bbase + ((kt + PIPE_) << 8));
    }
    SCHED_FENCE();
    acc = __builtin_amdgcn_wmma_f32_16x16x32_f16(
        false, a[cur].h, false, b[cur].h, (short)0, acc, false, false);
  }
  const int col   = nt * 16 + (lane & 15);
  const int rbase = rt * 16 + ((lane >> 4) << 3);
  #pragma unroll
  for (int r = 0; r < 8; ++r) raw[(size_t)(rbase + r) * HNP_ + col] = acc[r];
}

// ---------------------------------------------------------------------------
// MDN activations + scatter into concatenated outputs: pi|mu|sd|ro|eos.
// ---------------------------------------------------------------------------
__global__ void head_act_kernel(const float* __restrict__ raw,
    const float* __restrict__ pi_b, const float* __restrict__ mu_b,
    const float* __restrict__ sd_b, const float* __restrict__ ro_b,
    const float* __restrict__ eos_b, float* __restrict__ out) {
  int r = blockIdx.x * blockDim.x + threadIdx.x;
  if (r >= BT_) return;
  const float* v = raw + (size_t)r * HNP_;
  float* pi  = out;
  float* mu  = out + (size_t)BT_ * 20;
  float* sdo = out + (size_t)BT_ * 60;
  float* ro  = out + (size_t)BT_ * 100;
  float* eos = out + (size_t)BT_ * 120;

  float vv[20], mx = -3.4e38f;
  #pragma unroll
  for (int i = 0; i < 20; ++i) { vv[i] = v[i] + pi_b[i]; mx = fmaxf(mx, vv[i]); }
  float s = 0.0f;
  #pragma unroll
  for (int i = 0; i < 20; ++i) { float e = __expf(vv[i] - mx); vv[i] = e; s += e; }
  float inv = 1.0f / s;
  #pragma unroll
  for (int i = 0; i < 20; ++i) pi[(size_t)r * 20 + i] = vv[i] * inv;
  #pragma unroll
  for (int i = 0; i < 40; ++i) mu[(size_t)r * 40 + i]  = v[20 + i] + mu_b[i];
  #pragma unroll
  for (int i = 0; i < 40; ++i) sdo[(size_t)r * 40 + i] = __expf(v[60 + i] + sd_b[i]);
  #pragma unroll
  for (int i = 0; i < 20; ++i) ro[(size_t)r * 20 + i]  = tanhf(v[100 + i] + ro_b[i]);
  eos[r] = sigmoidf_(v[120] + eos_b[0]);
}

// ---------------------------------------------------------------------------
extern "C" void kernel_launch(void* const* d_in, const int* in_sizes, int n_in,
                              void* d_out, int out_size, void* d_ws, size_t ws_size,
                              hipStream_t stream) {
  (void)in_sizes; (void)n_in; (void)out_size; (void)ws_size;
  const float* x     = (const float*)d_in[0];
  const float* W_xh  = (const float*)d_in[1];
  const float* W_ci  = (const float*)d_in[2];
  const float* b_i   = (const float*)d_in[3];
  const float* W_cf  = (const float*)d_in[4];
  const float* b_f   = (const float*)d_in[5];
  const float* b_c   = (const float*)d_in[6];
  const float* W_co  = (const float*)d_in[7];
  const float* b_o   = (const float*)d_in[8];
  const float* pi_w  = (const float*)d_in[9];
  const float* pi_b  = (const float*)d_in[10];
  const float* mu_w  = (const float*)d_in[11];
  const float* mu_b  = (const float*)d_in[12];
  const float* sd_w  = (const float*)d_in[13];
  const float* sd_b  = (const float*)d_in[14];
  const float* ro_w  = (const float*)d_in[15];
  const float* ro_b  = (const float*)d_in[16];
  const float* eos_w = (const float*)d_in[17];
  const float* eos_b = (const float*)d_in[18];
  float* out = (float*)d_out;

  // Workspace carve-up (~163 MB total)
  char* ws = (char*)d_ws;
  size_t off = 0;
  auto take = [&](size_t bytes) -> char* {
    char* p = ws + off; off += (bytes + 255) & ~(size_t)255; return p;
  };
  unsigned* wf  = (unsigned*)take((size_t)NT_ * KT_ * 256 * 4);   // 6.7 MB  W_xh frags
  unsigned* hwf = (unsigned*)take((size_t)HNT_ * KT_ * 256 * 4);  // 0.24 MB head frags
  unsigned* af  = (unsigned*)take((size_t)4 * KT_ * 256 * 4);     // 119 KB  A frags
  float*    z   = (float*)take((size_t)B_ * K4_ * 4);             // 0.92 MB gates
  float*    c   = (float*)take((size_t)B_ * H_ * 4);              // 0.23 MB cell state
  unsigned* hsf = (unsigned*)take((size_t)RT_ * KT_ * 256 * 4);   // 121.6 MB hidden seq frags
  float*    raw = (float*)take((size_t)BT_ * HNP_ * 4);           // 33.5 MB head scores
  unsigned* cnt = (unsigned*)take(256);                           // barrier counter

  hipLaunchKernelGGL(prep_w_kernel,    dim3(NT_ * KT_),  dim3(256), 0, stream, W_xh, wf);
  hipLaunchKernelGGL(prep_head_kernel, dim3(HNT_ * KT_), dim3(256), 0, stream,
                     pi_w, mu_w, sd_w, ro_w, eos_w, hwf);
  hipLaunchKernelGGL(init_kernel,      dim3(2048),       dim3(256), 0, stream, af, c, hsf, cnt);
  hipLaunchKernelGGL(fill_x0_kernel,   dim3(1),          dim3(256), 0, stream, x, af);
  hipLaunchKernelGGL(lstm_persistent,  dim3(NT_),        dim3(128), 0, stream,
                     x, wf, W_ci, b_i, W_cf, b_f, b_c, W_co, b_o, af, z, c, hsf, cnt);
  hipLaunchKernelGGL(head_matmul_kernel, dim3(RT_),      dim3(256), 0, stream, hsf, hwf, raw);
  hipLaunchKernelGGL(head_act_kernel,  dim3((BT_ + 255) / 256), dim3(256), 0, stream,
                     raw, pi_b, mu_b, sd_b, ro_b, eos_b, out);
}
